// Tacotron_67808943669739
// MI455X (gfx1250) — matrix operand order repaired and checked by reading, not compiled, not run on hardware
//
#include <hip/hip_runtime.h>
#include <hip/hip_bf16.h>
#include <math.h>

// ---------------- CDNA5 WMMA types ----------------
typedef __attribute__((ext_vector_type(16))) _Float16 v16h;
typedef __attribute__((ext_vector_type(8)))  _Float16 v8h;
typedef __attribute__((ext_vector_type(8)))  float    v8f;
typedef __attribute__((ext_vector_type(4)))  float    v4f;

__device__ __forceinline__ float sigmf(float x) { return 1.f / (1.f + __expf(-x)); }

__device__ __forceinline__ v16h cat8(v8h a, v8h b) {
  return __builtin_shufflevector(a, b, 0,1,2,3,4,5,6,7,8,9,10,11,12,13,14,15);
}

// =====================================================================
// GEMM: C[M,N] = act( A[M,K] @ W[N,K]^T + bias[N] )
// A: f32 row-major. W: f16 row-major [N,K]. 2 N-tiles per wave (A reuse),
// 4 waves/block. Main K loop: unconditional vector loads (K multiple of 32);
// tail chunk uses clamped indices + v_cndmask (branch-free).
// act: 0=none 1=relu 2=sigmoid 3=tanh
// =====================================================================
__global__ void k_gemm(const float* __restrict__ A, int lda,
                       const _Float16* __restrict__ W, int ldw,
                       const float* __restrict__ bias,
                       float* __restrict__ C, long ldc,
                       int M, int N, int K, int act)
{
  const int wave = threadIdx.x >> 5;
  const int lane = threadIdx.x & 31;
  const int tn0 = (blockIdx.x * 4 + wave) * 2;       // first of 2 N tiles
  if (tn0 * 16 >= N) return;                         // wave-uniform guard
  const int tm = blockIdx.y;
  const int half = lane >> 4;
  const int m  = tm * 16 + (lane & 15);
  const int n0 = tn0 * 16 + (lane & 15);
  const int n1 = n0 + 16;
  const int n1c = (n1 < N) ? n1 : (N - 1);           // clamp (compute junk, guard store)

  const float*    __restrict__ Arow = A + (size_t)m * lda;
  const _Float16* __restrict__ W0   = W + (size_t)n0  * ldw;
  const _Float16* __restrict__ W1   = W + (size_t)n1c * ldw;

  v8f acc0, acc1;
  #pragma unroll
  for (int r = 0; r < 8; ++r) { acc0[r] = 0.f; acc1[r] = 0.f; }

  const int Kfull = K & ~31;
  for (int kk = 0; kk < Kfull; kk += 32) {
    // A fragment: K = kk + half*16 + i  -> 16 contiguous f32
    const v4f* a0p = (const v4f*)(Arow + kk + (half << 4));
    v4f a0 = a0p[0], a1 = a0p[1], a2 = a0p[2], a3 = a0p[3];
    v16h af;
    #pragma unroll
    for (int i = 0; i < 4; ++i) {
      af[i]      = (_Float16)a0[i];
      af[4 + i]  = (_Float16)a1[i];
      af[8 + i]  = (_Float16)a2[i];
      af[12 + i] = (_Float16)a3[i];
    }
    // B fragments: K = kk + half*16 + i -> 16 contiguous f16
    const v8h* b0p = (const v8h*)(W0 + kk + (half << 4));
    const v8h* b1p = (const v8h*)(W1 + kk + (half << 4));
    v16h bf0 = cat8(b0p[0], b0p[1]);
    v16h bf1 = cat8(b1p[0], b1p[1]);
    acc0 = __builtin_amdgcn_wmma_f32_16x16x32_f16(false, af, false, bf0, (short)0, acc0, false, false);
    acc1 = __builtin_amdgcn_wmma_f32_16x16x32_f16(false, af, false, bf1, (short)0, acc1, false, false);
  }
  if (Kfull < K) {  // K remainder (e.g. K=80): clamped loads, data select
    v16h af, bf0, bf1;
    #pragma unroll
    for (int i = 0; i < 16; ++i) {
      int ka  = Kfull + (half << 4) + i;
      float av = Arow[(ka < K) ? ka : (K - 1)];
      af[i] = (_Float16)((ka < K) ? av : 0.f);
      int kb  = Kfull + (half << 4) + i;
      int kbc = (kb < K) ? kb : (K - 1);
      _Float16 b0v = W0[kbc], b1v = W1[kbc];
      bf0[i] = (kb < K) ? b0v : (_Float16)0.f;
      bf1[i] = (kb < K) ? b1v : (_Float16)0.f;
    }
    acc0 = __builtin_amdgcn_wmma_f32_16x16x32_f16(false, af, false, bf0, (short)0, acc0, false, false);
    acc1 = __builtin_amdgcn_wmma_f32_16x16x32_f16(false, af, false, bf1, (short)0, acc1, false, false);
  }

  const float bv0 = (bias != nullptr) ? bias[n0] : 0.f;
  const float bv1 = (bias != nullptr && n1 < N) ? bias[n1] : 0.f;
  #pragma unroll
  for (int r = 0; r < 8; ++r) {
    int row = tm * 16 + half * 8 + r;               // C/D: VGPR r -> M = r + half*8
    float v0 = acc0[r] + bv0;
    float v1 = acc1[r] + bv1;
    if (act == 1) { v0 = fmaxf(v0, 0.f); v1 = fmaxf(v1, 0.f); }
    else if (act == 2) { v0 = sigmf(v0); v1 = sigmf(v1); }
    else if (act == 3) { v0 = tanhf(v0); v1 = tanhf(v1); }
    C[(size_t)row * ldc + n0] = v0;
    if (n1 < N) C[(size_t)row * ldc + n1] = v1;
  }
}

// =====================================================================
// Direct conv1d + fused inference BN (+optional relu) via WMMA.
// X: [B,T,Cin] f32 (channels contiguous).  Wt: f16 [ktap][Cout][Cin]
// (pre-transposed so A-fragments are contiguous f16 runs).
// 2 Cout-tiles per wave (activation B-fragment reuse); Cout%32==0.
// Time padding: clamp address, select data (branch-free).
// =====================================================================
__global__ void k_conv(const float* __restrict__ X, int Cin, int T,
                       const _Float16* __restrict__ Wt, int ktap, int padL, int Cout,
                       const float* __restrict__ gamma, const float* __restrict__ beta,
                       float* __restrict__ Out, int ldo, int chOff,
                       int doRelu, float bnInv)
{
  const int wave = threadIdx.x >> 5;
  const int lane = threadIdx.x & 31;
  const int tn = blockIdx.x * 4 + wave;              // time tile
  if (tn * 16 >= T) return;
  const int tm0 = blockIdx.y * 2;                    // two cout tiles
  const int b  = blockIdx.z;
  const int half = lane >> 4;
  const int tcol  = tn * 16 + (lane & 15);           // output time (N)
  const int co_a0 = tm0 * 16 + (lane & 15);          // weight rows (M)
  const int co_a1 = co_a0 + 16;

  v8f acc0, acc1;
  #pragma unroll
  for (int r = 0; r < 8; ++r) { acc0[r] = 0.f; acc1[r] = 0.f; }

  for (int dt = 0; dt < ktap; ++dt) {
    const int tin = tcol + dt - padL;
    const bool inb = (tin >= 0) && (tin < T);
    const int tcl = inb ? tin : 0;                   // clamp address only
    const float*    __restrict__ xrow  = X + ((size_t)b * T + tcl) * Cin;
    const _Float16* __restrict__ w0row = Wt + ((size_t)dt * Cout + co_a0) * Cin;
    const _Float16* __restrict__ w1row = Wt + ((size_t)dt * Cout + co_a1) * Cin;
    for (int kk = 0; kk < Cin; kk += 32) {
      // B fragment from activations: 16 contiguous f32 at kk + half*16
      const v4f* xp = (const v4f*)(xrow + kk + (half << 4));
      v4f x0 = xp[0], x1 = xp[1], x2 = xp[2], x3 = xp[3];
      v16h bf;
      #pragma unroll
      for (int i = 0; i < 4; ++i) {
        bf[i]      = (_Float16)(inb ? x0[i] : 0.f);  // v_cndmask, no branch
        bf[4 + i]  = (_Float16)(inb ? x1[i] : 0.f);
        bf[8 + i]  = (_Float16)(inb ? x2[i] : 0.f);
        bf[12 + i] = (_Float16)(inb ? x3[i] : 0.f);
      }
      // A fragments: 16 contiguous f16 at kk + half*16
      const v8h* w0a = (const v8h*)(w0row + kk + (half << 4));
      const v8h* w1a = (const v8h*)(w1row + kk + (half << 4));
      v16h af0 = cat8(w0a[0], w0a[1]);
      v16h af1 = cat8(w1a[0], w1a[1]);
      acc0 = __builtin_amdgcn_wmma_f32_16x16x32_f16(false, af0, false, bf, (short)0, acc0, false, false);
      acc1 = __builtin_amdgcn_wmma_f32_16x16x32_f16(false, af1, false, bf, (short)0, acc1, false, false);
    }
  }
  #pragma unroll
  for (int r = 0; r < 8; ++r) {
    int co0 = tm0 * 16 + half * 8 + r;
    int co1 = co0 + 16;
    float v0 = acc0[r] * (bnInv * gamma[co0]) + beta[co0];
    float v1 = acc1[r] * (bnInv * gamma[co1]) + beta[co1];
    if (doRelu) { v0 = fmaxf(v0, 0.f); v1 = fmaxf(v1, 0.f); }
    Out[((size_t)b * T + tcol) * ldo + chOff + co0] = v0;
    Out[((size_t)b * T + tcol) * ldo + chOff + co1] = v1;
  }
}

// ---------------- weight conversion (once per launch) ----------------
__global__ void k_cvt(const float* __restrict__ in, _Float16* __restrict__ out, long n)
{
  long i = (long)blockIdx.x * blockDim.x + threadIdx.x;
  if (i < n) out[i] = (_Float16)in[i];
}

// [Cout,Cin,ktap] f32 -> [ktap][Cout][Cin] f16
__global__ void k_cvt_conv(const float* __restrict__ in, _Float16* __restrict__ out,
                           int Cout, int Cin, int ktap)
{
  long idx = (long)blockIdx.x * blockDim.x + threadIdx.x;
  long total = (long)Cout * Cin * ktap;
  if (idx >= total) return;
  int co = (int)(idx / ((long)Cin * ktap));
  long rem = idx - (long)co * Cin * ktap;
  int ci = (int)(rem / ktap);
  int dt = (int)(rem - (long)ci * ktap);
  out[((long)dt * Cout + co) * Cin + ci] = (_Float16)in[idx];
}

// ---------------- small elementwise / reduction kernels ----------------
__global__ void k_embed(const int* __restrict__ ids, const float* __restrict__ emb,
                        float* __restrict__ out, int rows, int D)
{
  long i = (long)blockIdx.x * blockDim.x + threadIdx.x;
  long n = (long)rows * D;
  if (i >= n) return;
  long row = i / D, d = i - row * D;
  out[i] = emb[(long)ids[row] * D + d];
}

__global__ void k_fill(float* p, int n, float v)
{
  int i = blockIdx.x * blockDim.x + threadIdx.x;
  if (i < n) p[i] = v;
}

__global__ void k_onehot(float* w, int n, int Te)
{
  int i = blockIdx.x * blockDim.x + threadIdx.x;
  if (i < n) w[i] = ((i % Te) == 0) ? 1.f : 0.f;
}

__global__ void k_add(const float* a, const float* b, float* c, int n)
{
  int i = blockIdx.x * blockDim.x + threadIdx.x;
  if (i < n) c[i] = a[i] + b[i];
}

__global__ void k_hw_combine(const float* __restrict__ Hv, const float* __restrict__ Tv,
                             float* __restrict__ h, int n)
{
  int i = blockIdx.x * blockDim.x + threadIdx.x;
  if (i < n) { float t = Tv[i]; h[i] = Hv[i] * t + (1.f - t) * h[i]; }
}

// GRU gate combine: gi/gh already include biases (folded into GEMMs).
__global__ void k_gru_gate(const float* __restrict__ gi, long gi_rstride,
                           const float* __restrict__ gh,
                           float* __restrict__ hstate,
                           float* __restrict__ hout, long hout_rstride,
                           int B, int H)
{
  int idx = blockIdx.x * blockDim.x + threadIdx.x;
  if (idx >= B * H) return;
  int b = idx / H, j = idx - b * H;
  const float* gib = gi + (long)b * gi_rstride;
  const float* ghb = gh + (long)b * 3 * H;
  float r = sigmf(gib[j]         + ghb[j]);
  float z = sigmf(gib[H + j]     + ghb[H + j]);
  float n = tanhf(gib[2 * H + j] + r * ghb[2 * H + j]);
  float hv = (1.f - z) * n + z * hstate[idx];
  hstate[idx] = hv;
  if (hout) hout[(long)b * hout_rstride + j] = hv;
}

__global__ void k_concat2(float* __restrict__ out,
                          const float* __restrict__ a, long a_rstride, int Da,
                          const float* __restrict__ bsrc, int Db, int B)
{
  int idx = blockIdx.x * blockDim.x + threadIdx.x;
  int D = Da + Db;
  if (idx >= B * D) return;
  int b = idx / D, j = idx - b * D;
  out[idx] = (j < Da) ? a[(long)b * a_rstride + j] : bsrc[b * Db + (j - Da)];
}

// location conv: watt[B,Te] -> loc[(b*Te+t)*32 + co], k=31, pad 15
__global__ void k_loc_conv(const float* __restrict__ watt, const float* __restrict__ wl,
                           float* __restrict__ loc, int B, int Te)
{
  int idx = blockIdx.x * blockDim.x + threadIdx.x;
  if (idx >= B * Te * 32) return;
  int co = idx & 31, bt = idx >> 5;
  int b = bt / Te, t = bt - b * Te;
  float acc = 0.f;
  for (int dt = 0; dt < 31; ++dt) {
    int ti = t + dt - 15;
    bool ok = (ti >= 0) && (ti < Te);
    float wv = watt[b * Te + (ok ? ti : 0)];
    acc += wl[co * 31 + dt] * (ok ? wv : 0.f);
  }
  loc[idx] = acc;
}

// e[b,t] = sum_d tanh(enc_proj + dec_proj + loc)*eW[d] + eb   (wave reduce)
__global__ void k_energy(const float* __restrict__ ep, const float* __restrict__ dp,
                         const float* __restrict__ lp, const float* __restrict__ eW,
                         const float* __restrict__ eb, float* __restrict__ e,
                         int B, int Te)
{
  int lane = threadIdx.x & 31;
  int wv = (int)((blockIdx.x * (long)blockDim.x + threadIdx.x) >> 5);
  if (wv >= B * Te) return;
  int b = wv / Te;
  const float* eprow = ep + (size_t)wv * 128;
  const float* lprow = lp + (size_t)wv * 128;
  const float* dprow = dp + (size_t)b * 128;
  float s = 0.f;
  #pragma unroll
  for (int i = 0; i < 4; ++i) {
    int d = lane * 4 + i;
    s += tanhf(eprow[d] + dprow[d] + lprow[d]) * eW[d];
  }
  #pragma unroll
  for (int off = 16; off; off >>= 1) s += __shfl_xor(s, off, 32);
  if (lane == 0) e[wv] = s + eb[0];
}

__global__ void k_softmax(const float* __restrict__ e, float* __restrict__ w, int Te)
{
  __shared__ float red[256];
  int b = blockIdx.x, t = threadIdx.x;
  float v = e[b * Te + t];
  red[t] = v; __syncthreads();
  for (int s = 128; s; s >>= 1) { if (t < s) red[t] = fmaxf(red[t], red[t + s]); __syncthreads(); }
  float mx = red[0]; __syncthreads();
  float ex = __expf(v - mx);
  red[t] = ex; __syncthreads();
  for (int s = 128; s; s >>= 1) { if (t < s) red[t] += red[t + s]; __syncthreads(); }
  w[b * Te + t] = ex / red[0];
}

__global__ void k_ctx(const float* __restrict__ w, const float* __restrict__ enc,
                      float* __restrict__ ctx, int Te, int D)
{
  int b = blockIdx.x, d = threadIdx.x;
  float acc = 0.f;
  for (int t = 0; t < Te; ++t)
    acc += w[b * Te + t] * enc[((size_t)b * Te + t) * D + d];
  ctx[b * D + d] = acc;
}

__global__ void k_addln(const float* __restrict__ h1, const float* __restrict__ h2,
                        const float* __restrict__ g, const float* __restrict__ bt,
                        float* __restrict__ out, int H)
{
  __shared__ float red[256];
  int b = blockIdx.x, j = threadIdx.x;
  float v = h1[b * H + j] + h2[b * H + j];
  red[j] = v; __syncthreads();
  for (int s = 128; s; s >>= 1) { if (j < s) red[j] += red[j + s]; __syncthreads(); }
  float mu = red[0] / (float)H; __syncthreads();
  float d = v - mu;
  red[j] = d * d; __syncthreads();
  for (int s = 128; s; s >>= 1) { if (j < s) red[j] += red[j + s]; __syncthreads(); }
  float var = red[0] / (float)H;
  out[b * H + j] = d * rsqrtf(var + 1e-5f) * g[j] + bt[j];
}

// =====================================================================
// Host orchestration
// =====================================================================
static inline void gemm(hipStream_t s, const float* A, int lda, const _Float16* W, int ldw,
                        const float* bias, float* C, long ldc, int M, int N, int K, int act)
{
  dim3 g((unsigned)((N / 16 + 7) / 8), (unsigned)(M / 16), 1);
  k_gemm<<<g, 128, 0, s>>>(A, lda, W, ldw, bias, C, ldc, M, N, K, act);
}

extern "C" void kernel_launch(void* const* d_in, const int* in_sizes, int n_in,
                              void* d_out, int out_size, void* d_ws, size_t ws_size,
                              hipStream_t stream)
{
  (void)in_sizes; (void)n_in; (void)out_size; (void)ws_size;
  const int B = 32, Te = 256, Td = 200;
  const float BN_INV = 0.99999500003749972f; // 1/sqrt(1+1e-5)

  const int*   text_ids = (const int*)d_in[0];
  const float* mel_in   = (const float*)d_in[1];
  auto F = [&](int i) -> const float* { return (const float*)d_in[i]; };

  // --- params flattened in setup_inputs() dict-insertion order ---
  const float* embedW = F(2);
  const float* epb1 = F(4); const float* epb2 = F(6);
  const float* bn1_g = F(56); const float* bn1_b = F(57);
  const float* bn2_g = F(59); const float* bn2_b = F(60);
  const float* hwHb = F(62); const float* hwTb = F(64);
  const float* fw_bih = F(67); const float* fw_bhh = F(68);
  const float* bw_bih = F(71); const float* bw_bhh = F(72);
  const float* dpb1 = F(74); const float* dpb2 = F(76);
  const float* ar_bih = F(79); const float* ar_bhh = F(80);
  const float* loc_cw = F(83);
  const float* eW = F(85); const float* eb = F(86);
  const float* g1_bih = F(89); const float* g1_bhh = F(90);
  const float* g2_bih = F(93); const float* g2_bhh = F(94);
  const float* ln_g = F(95); const float* ln_b = F(96);
  const float* mel_b = F(98);

  // --- workspace bump allocator ---
  char* wp = (char*)d_ws;
  auto allocf = [&](size_t nf) -> float* {
    float* r = (float*)wp;
    wp += ((nf * sizeof(float) + 255) / 256) * 256;
    return r;
  };
  auto alloch = [&](size_t nh) -> _Float16* {
    _Float16* r = (_Float16*)wp;
    wp += ((nh * sizeof(_Float16) + 255) / 256) * 256;
    return r;
  };
  auto cvt = [&](const float* src, _Float16* dst, long n) {
    k_cvt<<<(unsigned)((n + 255) / 256), 256, 0, stream>>>(src, dst, n);
  };

  // f32 activations
  float* emb    = allocf((size_t)B * Te * 256);
  float* t256   = allocf((size_t)B * Te * 256);
  float* x0     = allocf((size_t)B * Te * 128);
  float* bank   = allocf((size_t)B * Te * 2048);
  float* p1b    = allocf((size_t)B * Te * 128);
  float* hbuf   = allocf((size_t)B * Te * 128);
  float* Hb     = allocf((size_t)B * Te * 128);
  float* Tb     = allocf((size_t)B * Te * 128);
  float* gi_fw  = allocf((size_t)B * Te * 192);
  float* gi_bw  = allocf((size_t)B * Te * 192);
  float* ghe    = allocf((size_t)B * 192);
  float* hfw    = allocf((size_t)B * 64);
  float* hbw    = allocf((size_t)B * 64);
  float* enc_out  = allocf((size_t)B * Te * 128);
  float* enc_proj = allocf((size_t)B * Te * 128);
  float* dt256  = allocf((size_t)B * Td * 256);
  float* p_all  = allocf((size_t)B * Td * 128);
  float* attn_h = allocf((size_t)B * 256);
  float* h1     = allocf((size_t)B * 256);
  float* h2     = allocf((size_t)B * 256);
  float* ctx    = allocf((size_t)B * 128);
  float* watt   = allocf((size_t)B * Te);
  float* xcat   = allocf((size_t)B * 256);
  float* gib    = allocf((size_t)B * 768);
  float* ghb    = allocf((size_t)B * 768);
  float* dproj  = allocf((size_t)B * 128);
  float* loc    = allocf((size_t)B * Te * 32);
  float* locp   = allocf((size_t)B * Te * 128);
  float* ebuf   = allocf((size_t)B * Te);
  float* sbuf   = allocf((size_t)B * 256);

  // f16 weights (converted once per launch)
  _Float16* epW1h = alloch(256 * 256);  cvt(F(3), epW1h, 256 * 256);
  _Float16* epW2h = alloch(128 * 256);  cvt(F(5), epW2h, 128 * 256);
  _Float16* bankh[16];
  for (int k = 1; k <= 16; ++k) {
    long n = (long)128 * 128 * k;
    bankh[k - 1] = alloch(n);
    k_cvt_conv<<<(unsigned)((n + 255) / 256), 256, 0, stream>>>(F(7 + (k - 1)), bankh[k - 1], 128, 128, k);
  }
  _Float16* proj1h = alloch((long)3 * 128 * 2048);
  k_cvt_conv<<<(unsigned)(((long)128 * 2048 * 3 + 255) / 256), 256, 0, stream>>>(F(55), proj1h, 128, 2048, 3);
  _Float16* proj2h = alloch((long)3 * 128 * 128);
  k_cvt_conv<<<(unsigned)(((long)128 * 128 * 3 + 255) / 256), 256, 0, stream>>>(F(58), proj2h, 128, 128, 3);
  _Float16* hwHWh = alloch(4 * 128 * 128); cvt(F(61), hwHWh, 4 * 128 * 128);
  _Float16* hwTWh = alloch(4 * 128 * 128); cvt(F(63), hwTWh, 4 * 128 * 128);
  _Float16* fwWihh = alloch(192 * 128); cvt(F(65), fwWihh, 192 * 128);
  _Float16* fwWhhh = alloch(192 * 64);  cvt(F(66), fwWhhh, 192 * 64);
  _Float16* bwWihh = alloch(192 * 128); cvt(F(69), bwWihh, 192 * 128);
  _Float16* bwWhhh = alloch(192 * 64);  cvt(F(70), bwWhhh, 192 * 64);
  _Float16* dpW1h = alloch(256 * 80);  cvt(F(73), dpW1h, 256 * 80);
  _Float16* dpW2h = alloch(128 * 256); cvt(F(75), dpW2h, 128 * 256);
  _Float16* arWihh = alloch(768 * 256); cvt(F(77), arWihh, 768 * 256);
  _Float16* arWhhh = alloch(768 * 256); cvt(F(78), arWhhh, 768 * 256);
  _Float16* encWh = alloch(128 * 128); cvt(F(81), encWh, 128 * 128);
  _Float16* decWh = alloch(128 * 256); cvt(F(82), decWh, 128 * 256);
  _Float16* locWh = alloch(128 * 32);  cvt(F(84), locWh, 128 * 32);
  _Float16* g1Wihh = alloch(768 * 256); cvt(F(87), g1Wihh, 768 * 256);
  _Float16* g1Whhh = alloch(768 * 256); cvt(F(88), g1Whhh, 768 * 256);
  _Float16* g2Wihh = alloch(768 * 256); cvt(F(91), g2Wihh, 768 * 256);
  _Float16* g2Whhh = alloch(768 * 256); cvt(F(92), g2Whhh, 768 * 256);
  _Float16* melWh = alloch(400 * 256); cvt(F(97), melWh, 400 * 256);

  const int MT = B * Te;   // 8192 rows
  const int MD = B * Td;   // 6400 rows

  // ================= Encoder =================
  k_embed<<<(unsigned)(((long)MT * 256 + 255) / 256), 256, 0, stream>>>(text_ids, embedW, emb, MT, 256);
  gemm(stream, emb, 256, epW1h, 256, epb1, t256, 256, MT, 256, 256, 1);
  gemm(stream, t256, 256, epW2h, 256, epb2, x0, 128, MT, 128, 256, 1);

  // conv bank (k=1..16) + BN + relu -> bank[B,Te,2048]
  for (int k = 1; k <= 16; ++k) {
    dim3 g(4, 4, B);   // (Te/16/4, Cout/32, B)
    k_conv<<<g, 128, 0, stream>>>(x0, 128, Te, bankh[k - 1], k, (k - 1) / 2, 128,
                                  F(23 + (k - 1)), F(39 + (k - 1)),
                                  bank, 2048, 128 * (k - 1), 1, BN_INV);
  }
  { dim3 g(4, 4, B);
    k_conv<<<g, 128, 0, stream>>>(bank, 2048, Te, proj1h, 3, 1, 128, bn1_g, bn1_b,
                                  p1b, 128, 0, 1, BN_INV);
    k_conv<<<g, 128, 0, stream>>>(p1b, 128, Te, proj2h, 3, 1, 128, bn2_g, bn2_b,
                                  hbuf, 128, 0, 0, BN_INV); }
  k_add<<<(MT * 128 + 255) / 256, 256, 0, stream>>>(hbuf, x0, hbuf, MT * 128);

  // highway x4 (in place on hbuf)
  for (int i = 0; i < 4; ++i) {
    gemm(stream, hbuf, 128, hwHWh + (size_t)i * 128 * 128, 128, hwHb + i * 128, Hb, 128, MT, 128, 128, 1);
    gemm(stream, hbuf, 128, hwTWh + (size_t)i * 128 * 128, 128, hwTb + i * 128, Tb, 128, MT, 128, 128, 2);
    k_hw_combine<<<(MT * 128 + 255) / 256, 256, 0, stream>>>(Hb, Tb, hbuf, MT * 128);
  }

  // bidirectional GRU: batched input gates, sequential recurrent steps
  gemm(stream, hbuf, 128, fwWihh, 128, fw_bih, gi_fw, 192, MT, 192, 128, 0);
  gemm(stream, hbuf, 128, bwWihh, 128, bw_bih, gi_bw, 192, MT, 192, 128, 0);
  k_fill<<<(B * 64 + 255) / 256, 256, 0, stream>>>(hfw, B * 64, 0.f);
  k_fill<<<(B * 64 + 255) / 256, 256, 0, stream>>>(hbw, B * 64, 0.f);
  for (int t = 0; t < Te; ++t) {
    gemm(stream, hfw, 64, fwWhhh, 64, fw_bhh, ghe, 192, B, 192, 64, 0);
    k_gru_gate<<<(B * 64 + 255) / 256, 256, 0, stream>>>(
        gi_fw + (size_t)t * 192, (long)Te * 192, ghe, hfw,
        enc_out + (size_t)t * 128, (long)Te * 128, B, 64);
  }
  for (int t = Te - 1; t >= 0; --t) {
    gemm(stream, hbw, 64, bwWhhh, 64, bw_bhh, ghe, 192, B, 192, 64, 0);
    k_gru_gate<<<(B * 64 + 255) / 256, 256, 0, stream>>>(
        gi_bw + (size_t)t * 192, (long)Te * 192, ghe, hbw,
        enc_out + (size_t)t * 128 + 64, (long)Te * 128, B, 64);
  }
  gemm(stream, enc_out, 128, encWh, 128, nullptr, enc_proj, 128, MT, 128, 128, 0);

  // ================= Decoder prenet (teacher forced, batched) =================
  gemm(stream, mel_in, 80, dpW1h, 80, dpb1, dt256, 256, MD, 256, 80, 1);
  gemm(stream, dt256, 256, dpW2h, 256, dpb2, p_all, 128, MD, 128, 256, 1);

  // ================= Decoder state init =================
  k_fill<<<(B * 256 + 255) / 256, 256, 0, stream>>>(attn_h, B * 256, 0.f);
  k_fill<<<(B * 256 + 255) / 256, 256, 0, stream>>>(h1, B * 256, 0.f);
  k_fill<<<(B * 256 + 255) / 256, 256, 0, stream>>>(h2, B * 256, 0.f);
  k_fill<<<(B * 128 + 255) / 256, 256, 0, stream>>>(ctx, B * 128, 0.f);
  k_onehot<<<(B * Te + 255) / 256, 256, 0, stream>>>(watt, B * Te, Te);

  float* mels = (float*)d_out;  // [B, Td, 400]

  // ================= Decoder loop (sequential, 200 steps) =================
  for (int t = 0; t < Td; ++t) {
    const float* p_t = p_all + (size_t)t * 128;   // rows at b*Td + t
    // attention RNN: x = [p_t, ctx]
    k_concat2<<<(B * 256 + 255) / 256, 256, 0, stream>>>(xcat, p_t, (long)Td * 128, 128, ctx, 128, B);
    gemm(stream, xcat, 256, arWihh, 256, ar_bih, gib, 768, B, 768, 256, 0);
    gemm(stream, attn_h, 256, arWhhh, 256, ar_bhh, ghb, 768, B, 768, 256, 0);
    k_gru_gate<<<(B * 256 + 255) / 256, 256, 0, stream>>>(gib, 768, ghb, attn_h, nullptr, 0, B, 256);
    // location-sensitive attention
    gemm(stream, attn_h, 256, decWh, 256, nullptr, dproj, 128, B, 128, 256, 0);
    k_loc_conv<<<(B * Te * 32 + 255) / 256, 256, 0, stream>>>(watt, loc_cw, loc, B, Te);
    gemm(stream, loc, 32, locWh, 32, nullptr, locp, 128, MT, 128, 32, 0);
    k_energy<<<(B * Te) / 4, 128, 0, stream>>>(enc_proj, dproj, locp, eW, eb, ebuf, B, Te);
    k_softmax<<<B, Te, 0, stream>>>(ebuf, watt, Te);
    k_ctx<<<B, 128, 0, stream>>>(watt, enc_out, ctx, Te, 128);
    // decoder RNNs
    k_concat2<<<(B * 256 + 255) / 256, 256, 0, stream>>>(xcat, p_t, (long)Td * 128, 128, ctx, 128, B);
    gemm(stream, xcat, 256, g1Wihh, 256, g1_bih, gib, 768, B, 768, 256, 0);
    gemm(stream, h1, 256, g1Whhh, 256, g1_bhh, ghb, 768, B, 768, 256, 0);
    k_gru_gate<<<(B * 256 + 255) / 256, 256, 0, stream>>>(gib, 768, ghb, h1, nullptr, 0, B, 256);
    gemm(stream, h1, 256, g2Wihh, 256, g2_bih, gib, 768, B, 768, 256, 0);
    gemm(stream, h2, 256, g2Whhh, 256, g2_bhh, ghb, 768, B, 768, 256, 0);
    k_gru_gate<<<(B * 256 + 255) / 256, 256, 0, stream>>>(gib, 768, ghb, h2, nullptr, 0, B, 256);
    // layernorm(h1+h2) -> mel frame
    k_addln<<<B, 256, 0, stream>>>(h1, h2, ln_g, ln_b, sbuf, 256);
    gemm(stream, sbuf, 256, melWh, 256, mel_b, mels + (size_t)t * 400, (long)Td * 400,
         B, 400, 256, 0);
  }
}
